// BitLinear_65687229825776
// MI455X (gfx1250) — 8-bit matrix operand order repaired and checked by reading, compiled, not run on hardware
//
#include <hip/hip_runtime.h>
#include <hip/hip_bf16.h>

typedef __attribute__((ext_vector_type(8))) int v8i;

#define M_DIM 8192
#define N_DIM 11008
#define K_DIM 4096


__device__ __forceinline__ void async_copy_b128(const void* lds_dst, const void* gsrc) {
  unsigned lds_off = (unsigned)(size_t)lds_dst;
  unsigned long long gaddr = (unsigned long long)(size_t)gsrc;
  asm volatile("global_load_async_to_lds_b128 %0, %1, off"
               :: "v"(lds_off), "v"(gaddr) : "memory");
}

__device__ __forceinline__ void wait_async_all() {
  asm volatile("s_wait_asynccnt 0" ::: "memory");
}


__global__ void init_scales_k(float* s) {
  if (threadIdx.x == 0) { s[0] = 0.f; s[1] = 0.f; }
}

__global__ void absmax_k(const float* __restrict__ x, int n, float* __restrict__ s) {
  float m = 0.f;
  for (int i = blockIdx.x * blockDim.x + threadIdx.x; i < n; i += gridDim.x * blockDim.x)
    m = fmaxf(m, fabsf(x[i]));
  __shared__ float red[256];
  red[threadIdx.x] = m;
  __syncthreads();
  for (int o = 128; o > 0; o >>= 1) {
    if (threadIdx.x < o) red[threadIdx.x] = fmaxf(red[threadIdx.x], red[threadIdx.x + o]);
    __syncthreads();
  }
  if (threadIdx.x == 0)
    atomicMax((unsigned int*)&s[0], __float_as_uint(red[0]));
}

__global__ void abssum_k(const float* __restrict__ w, int n, float* __restrict__ s) {
  float a = 0.f;
  for (int i = blockIdx.x * blockDim.x + threadIdx.x; i < n; i += gridDim.x * blockDim.x)
    a += fabsf(w[i]);
  __shared__ float red[256];
  red[threadIdx.x] = a;
  __syncthreads();
  for (int o = 128; o > 0; o >>= 1) {
    if (threadIdx.x < o) red[threadIdx.x] += red[threadIdx.x + o];
    __syncthreads();
  }
  if (threadIdx.x == 0)
    atomicAdd(&s[1], red[0]);
}


__device__ __forceinline__ int clamp_q(float f, float lim) {
  return (int)fminf(fmaxf(rintf(f), -lim), lim);
}

__device__ __forceinline__ int pack4(float a, float b, float c, float d, float inv, float lim) {
  return (clamp_q(a * inv, lim) & 255) |
         ((clamp_q(b * inv, lim) & 255) << 8) |
         ((clamp_q(c * inv, lim) & 255) << 16) |
         ((clamp_q(d * inv, lim) & 255) << 24);
}

__global__ void quant_x_k(const float4* __restrict__ x4, const float* __restrict__ s,
                          int* __restrict__ q, int n4) {
  int i = blockIdx.x * blockDim.x + threadIdx.x;
  if (i >= n4) return;
  float sc = fmaxf(s[0] * (1.f / 127.f), 1e-8f);
  float inv = 1.f / sc;
  float4 v = x4[i];
  q[i] = pack4(v.x, v.y, v.z, v.w, inv, 127.f);
}

__global__ void tern_w_k(const float4* __restrict__ w4, const float* __restrict__ s,
                         int* __restrict__ q, int n4) {
  int i = blockIdx.x * blockDim.x + threadIdx.x;
  if (i >= n4) return;
  float sc = s[1] * (1.f / ((float)N_DIM * (float)K_DIM)) + 1e-8f;
  float inv = 1.f / sc;
  float4 v = w4[i];
  q[i] = pack4(v.x, v.y, v.z, v.w, inv, 1.f);
}


#define LDS_PITCH 80

__device__ __forceinline__ v8i load_bfrag(const signed char* base, int row, int bhalf) {
  const signed char* p = base + row * LDS_PITCH + bhalf;
  union { v8i v; int4 q[2]; } u;
  u.q[0] = *(const int4*)(p);
  u.q[1] = *(const int4*)(p + 32);
  return u.v;
}

__device__ __forceinline__ v8i load_afrag(const signed char* base, int row, int ahalf) {
  const signed char* p = base + row * LDS_PITCH + ahalf;
  union { v8i v; long long q[4]; } u;
  u.q[0] = *(const long long*)(p);
  u.q[1] = *(const long long*)(p + 8);
  u.q[2] = *(const long long*)(p + 32);
  u.q[3] = *(const long long*)(p + 40);
  return u.v;
}

__device__ __forceinline__ v8i wmma_iu8(v8i a, v8i b, v8i c) {
  return __builtin_amdgcn_wmma_i32_16x16x64_iu8(true, a, true, b, c, false, false);
}

__device__ __forceinline__ void store_tile(float* __restrict__ out, int mrow, int ncol,
                                           v8i c, float sc) {
#pragma unroll
  for (int v = 0; v < 8; ++v)
    out[(long)(mrow + v) * N_DIM + ncol] = (float)c[v] * sc;
}

__launch_bounds__(256)
__global__ void bitlinear_gemm_k(const signed char* __restrict__ qx,
                                 const signed char* __restrict__ tw,
                                 const float* __restrict__ scales,
                                 float* __restrict__ out) {
  __shared__ __align__(16) signed char sA[2][128 * LDS_PITCH];
  __shared__ __align__(16) signed char sB[2][128 * LDS_PITCH];

  const int tid   = threadIdx.x;
  const int lane  = tid & 31;
  const int wave  = tid >> 5;
  const int wm    = (wave & 1) * 64;
  const int wn    = (wave >> 1) * 32;
  const int r16   = lane & 15;
  const int half  = lane >> 4;
  const int ahalf = half << 4;
  const int bhalf = half << 4;

  const long aBase = (long)blockIdx.y * 128 * K_DIM;
  const long bBase = (long)blockIdx.x * 128 * K_DIM;

  const int r0  = tid >> 2,          kc = (tid & 3) << 4;
  const int r1  = (tid + 256) >> 2;
  const int l0  = r0 * LDS_PITCH + kc;
  const int l1  = r1 * LDS_PITCH + kc;
  const long g0 = (long)r0 * K_DIM + kc;
  const long g1 = (long)r1 * K_DIM + kc;

  const signed char* curA = sA[0];
  const signed char* curB = sB[0];
  const signed char* nxtA = sA[1];
  const signed char* nxtB = sB[1];

  async_copy_b128(curA + l0, qx + aBase + g0);
  async_copy_b128(curA + l1, qx + aBase + g1);
  async_copy_b128(curB + l0, tw + bBase + g0);
  async_copy_b128(curB + l1, tw + bBase + g1);

  v8i acc00 = {}, acc01 = {}, acc10 = {}, acc11 = {};
  v8i acc20 = {}, acc21 = {}, acc30 = {}, acc31 = {};

  const int NK = K_DIM / 64;
#pragma unroll 1
  for (int kt = 0; kt < NK; ++kt) {
    wait_async_all();
    __syncthreads();
    if (kt + 1 < NK) {
      const long k0 = (long)(kt + 1) * 64;
      async_copy_b128(nxtA + l0, qx + aBase + g0 + k0);
      async_copy_b128(nxtA + l1, qx + aBase + g1 + k0);
      async_copy_b128(nxtB + l0, tw + bBase + g0 + k0);
      async_copy_b128(nxtB + l1, tw + bBase + g1 + k0);
    }

    const v8i b0 = load_bfrag(curB, wn + r16, bhalf);
    const v8i b1 = load_bfrag(curB, wn + 16 + r16, bhalf);
    const v8i a0 = load_afrag(curA, wm + r16, ahalf);
    const v8i a1 = load_afrag(curA, wm + 16 + r16, ahalf);
    const v8i a2 = load_afrag(curA, wm + 32 + r16, ahalf);
    const v8i a3 = load_afrag(curA, wm + 48 + r16, ahalf);

    acc00 = wmma_iu8(a0, b0, acc00);
    acc01 = wmma_iu8(a0, b1, acc01);
    acc10 = wmma_iu8(a1, b0, acc10);
    acc11 = wmma_iu8(a1, b1, acc11);
    acc20 = wmma_iu8(a2, b0, acc20);
    acc21 = wmma_iu8(a2, b1, acc21);
    acc30 = wmma_iu8(a3, b0, acc30);
    acc31 = wmma_iu8(a3, b1, acc31);

    const signed char* t;
    t = curA; curA = nxtA; nxtA = t;
    t = curB; curB = nxtB; nxtB = t;
  }

  const float act_scale = fmaxf(scales[0] * (1.f / 127.f), 1e-8f);
  const float w_scale   = scales[1] * (1.f / ((float)N_DIM * (float)K_DIM)) + 1e-8f;
  const float sc = act_scale * w_scale;

  const int mb = blockIdx.y * 128 + wm + (half << 3);
  const int nb = blockIdx.x * 128 + wn + r16;

  store_tile(out, mb +  0, nb +  0, acc00, sc);
  store_tile(out, mb +  0, nb + 16, acc01, sc);
  store_tile(out, mb + 16, nb +  0, acc10, sc);
  store_tile(out, mb + 16, nb + 16, acc11, sc);
  store_tile(out, mb + 32, nb +  0, acc20, sc);
  store_tile(out, mb + 32, nb + 16, acc21, sc);
  store_tile(out, mb + 48, nb +  0, acc30, sc);
  store_tile(out, mb + 48, nb + 16, acc31, sc);
}


extern "C" void kernel_launch(void* const* d_in, const int* in_sizes, int n_in,
                              void* d_out, int out_size, void* d_ws, size_t ws_size,
                              hipStream_t stream) {
  const float* x = (const float*)d_in[0];
  const float* W = (const float*)d_in[1];
  float* out = (float*)d_out;

  float* scales   = (float*)d_ws;
  signed char* qx = (signed char*)d_ws + 256;
  signed char* tw = qx + (size_t)M_DIM * K_DIM;

  const int nx = M_DIM * K_DIM;
  const int nw = N_DIM * K_DIM;

  init_scales_k<<<1, 64, 0, stream>>>(scales);
  absmax_k<<<4096, 256, 0, stream>>>(x, nx, scales);
  abssum_k<<<4096, 256, 0, stream>>>(W, nw, scales);
  quant_x_k<<<(nx / 4 + 255) / 256, 256, 0, stream>>>((const float4*)x, scales, (int*)qx, nx / 4);
  tern_w_k<<<(nw / 4 + 255) / 256, 256, 0, stream>>>((const float4*)W, scales, (int*)tw, nw / 4);

  dim3 grid(N_DIM / 128, M_DIM / 128);
  bitlinear_gemm_k<<<grid, 256, 0, stream>>>(qx, tw, scales, out);
}
